// DissipativeHebbianLayer_25108378812947
// MI455X (gfx1250) — matrix operand order repaired and checked by reading, not compiled, not run on hardware
//
#include <hip/hip_runtime.h>
#include <hip/hip_bf16.h>

#define ETA   0.1f
#define Bsz   4
#define Nseq  2048
#define Dm    1024
#define Hh    8
#define DHd   64
#define HD    (Hh * DHd)     // 512
#define CT    64             // chunk length
#define NC    (Nseq / CT)    // 32

typedef __attribute__((ext_vector_type(16))) __bf16 v16bf;
typedef __attribute__((ext_vector_type(8)))  float  v8f;
typedef __attribute__((ext_vector_type(2)))  float  v2f;

__device__ inline v8f wmma_f32x4(v2f a, v2f b, v8f c) {
    return __builtin_amdgcn_wmma_f32_16x16x4_f32(false, a, false, b, (short)0, c, false, false);
}
__device__ inline v8f wmma_bf16(v16bf a, v16bf b, v8f c) {
    return __builtin_amdgcn_wmma_f32_16x16x32_bf16(false, a, false, b, (short)0, c, false, false);
}

// async global->LDS b128 copy (ASYNCcnt path, gfx1250)
__device__ inline void async_ld_b128(void* lds_ptr, const void* gptr) {
    unsigned la = (unsigned)(unsigned long long)lds_ptr;          // LDS offset = addr[31:0]
    unsigned long long ga = (unsigned long long)gptr;
    asm volatile("global_load_async_to_lds_b128 %0, %1, off" : : "v"(la), "v"(ga) : "memory");
}
__device__ inline void async_wait0() {
    asm volatile("s_wait_asynccnt 0x0" : : : "memory");
}

// ---------------- decay cumulative products (per batch, per chunk) ------------
__global__ void decay_cum_kernel(const float* __restrict__ gamma, float* __restrict__ cum) {
    int idx = blockIdx.x * blockDim.x + threadIdx.x;
    if (idx >= Bsz * NC) return;
    int b = idx / NC, c = idx % NC;
    float p = 1.0f;
    for (int t = 0; t < CT; ++t) {
        int n = c * CT + t;
        float g = fminf(gamma[b * Nseq + n], 10.0f);   // DT = 1
        p *= expf(-g);
        cum[b * Nseq + n] = p;
    }
}

// ---------------- f32 -> bf16 row-major pack (for A fragments) ----------------
__global__ void pack_bf16_kernel(const float* __restrict__ src, __bf16* __restrict__ dst, int n4) {
    int tid = blockIdx.x * blockDim.x + threadIdx.x;
    if (tid >= n4) return;
    float4 f = ((const float4*)src)[tid];
    union { __bf16 h[4]; unsigned long long u; } b;
    b.h[0] = (__bf16)f.x; b.h[1] = (__bf16)f.y; b.h[2] = (__bf16)f.z; b.h[3] = (__bf16)f.w;
    ((unsigned long long*)dst)[tid] = b.u;
}

// ---------------- weight pack into WMMA B-fragment order ----------------------
// PW[((nt*KB + kb)*32 + lane)*16 + i] = bf16( W[(kb*32 + (lane>>4)*16 + i)*Nout + nt*16 + (lane&15)] )
__global__ void pack_w_kernel(const float* __restrict__ W, __bf16* __restrict__ PW,
                              int Kdim, int Nout) {
    int tid = blockIdx.x * blockDim.x + threadIdx.x;
    int KB = Kdim >> 5;
    int total = (Nout >> 4) * KB * 32;
    if (tid >= total) return;
    int lane = tid & 31;
    int kb = (tid >> 5) % KB;
    int nt = (tid >> 5) / KB;
    int hh = lane >> 4, r = lane & 15;
    union { __bf16 h[16]; uint4 u[2]; } buf;
#pragma unroll
    for (int i = 0; i < 16; ++i)
        buf.h[i] = (__bf16)W[(size_t)(kb * 32 + hh * 16 + i) * Nout + nt * 16 + r];
    uint4* dst = (uint4*)(PW + (size_t)tid * 16);
    dst[0] = buf.u[0];
    dst[1] = buf.u[1];
}

// ---------------- generic packed-operand bf16 WMMA GEMM core ------------------
__device__ inline void gemm_tile_bf16(const __bf16* __restrict__ arow,   // row m, Kdim elems
                                      const __bf16* __restrict__ bptr,   // packed B frags
                                      int KB, int hh, v8f& acc) {
    for (int kb = 0; kb < KB; ++kb) {
        union { uint4 u[2]; v16bf v; } A, B;
        A.u[0] = *(const uint4*)(arow + kb * 32 + hh * 16);
        A.u[1] = *(const uint4*)(arow + kb * 32 + hh * 16 + 8);
        const uint4* bp = (const uint4*)(bptr + (size_t)kb * 32 * 16);
        B.u[0] = bp[0];
        B.u[1] = bp[1];
        __builtin_prefetch(bptr + (size_t)(kb + 2) * 32 * 16, 0, 1);
        acc = wmma_bf16(A.v, B.v, acc);
    }
}

// ---------------- q/k/v projections: bf16 WMMA GEMM ---------------------------
__global__ void __launch_bounds__(128) proj_kernel(
    const __bf16* __restrict__ xb,
    const __bf16* __restrict__ PWq, const __bf16* __restrict__ PWk, const __bf16* __restrict__ PWv,
    float* __restrict__ q, float* __restrict__ k, float* __restrict__ v) {
    const __bf16* PW = (blockIdx.z == 0) ? PWq : (blockIdx.z == 1) ? PWk : PWv;
    float* outp     = (blockIdx.z == 0) ? q   : (blockIdx.z == 1) ? k   : v;
    int lane = threadIdx.x & 31, wv = threadIdx.x >> 5;
    int r = lane & 15, hh = lane >> 4;
    const int KB = Dm >> 5;                         // 32
    int m  = blockIdx.x * 16 + r;
    int nt = blockIdx.y * 4 + wv;
    int col = nt * 16 + r;
    v8f acc = {};
    gemm_tile_bf16(xb + (size_t)m * Dm,
                   PW + ((size_t)(nt * KB) * 32 + lane) * 16, KB, hh, acc);
#pragma unroll
    for (int i = 0; i < 8; ++i) {
        int mm = blockIdx.x * 16 + i + 8 * hh;
        outp[(size_t)mm * HD + col] = acc[i];
    }
}

// ---------------- chunk-parallel pass: S=QV^T, Y_intra=(M.S)K, G=K'^T V -------
__global__ void __launch_bounds__(256) chunk_intra_kernel(
    const float* __restrict__ q, const float* __restrict__ k, const float* __restrict__ v,
    const float* __restrict__ cum, float* __restrict__ yws, float* __restrict__ Gws) {
    __shared__ float Qs[CT][DHd];
    __shared__ float Ks[CT][DHd];
    __shared__ float Vs[CT][DHd];
    __shared__ float cs[CT];
    float (*Ss)[DHd] = Qs;   // S overwrites Q's LDS after S is computed

    int c = blockIdx.x, h = blockIdx.y, b = blockIdx.z;
    int lane = threadIdx.x & 31, wv = threadIdx.x >> 5;
    int r = lane & 15, hh = lane >> 4;

    const size_t rowbase = ((size_t)(b * Nseq + c * CT)) * HD + h * DHd;
    for (int i4 = threadIdx.x; i4 < (CT * DHd) / 4; i4 += 256) {
        int t = i4 >> 4, i = (i4 & 15) * 4;
        size_t g = rowbase + (size_t)t * HD + i;
        async_ld_b128(&Qs[t][i], q + g);
        async_ld_b128(&Ks[t][i], k + g);
        async_ld_b128(&Vs[t][i], v + g);
    }
    if (threadIdx.x < CT) cs[threadIdx.x] = cum[b * Nseq + c * CT + threadIdx.x];
    async_wait0();
    __syncthreads();
    float P = cs[CT - 1];

    // S[t][tau] = sum_j Q[t][j] * V[tau][j]   (fp32 WMMA)
    v8f accS[2];
#pragma unroll
    for (int p = 0; p < 2; ++p) {
        int tt = wv + p * 8, mt = tt >> 2, nt = tt & 3;
        v8f acc = {};
        for (int k0 = 0; k0 < DHd; k0 += 4) {
            v2f a, bb;
#pragma unroll
            for (int e = 0; e < 2; ++e) {
                int kk = k0 + e + 2 * hh;
                a[e]  = Qs[mt * 16 + r][kk];
                bb[e] = Vs[nt * 16 + r][kk];
            }
            acc = wmma_f32x4(a, bb, acc);
        }
        accS[p] = acc;
    }
    __syncthreads();
#pragma unroll
    for (int p = 0; p < 2; ++p) {
        int tt = wv + p * 8, mt = tt >> 2, nt = tt & 3;
#pragma unroll
        for (int i = 0; i < 8; ++i) Ss[mt * 16 + i + 8 * hh][nt * 16 + r] = accS[p][i];
    }
    __syncthreads();

    // Y_intra[t][i] = sum_{tau<=t} eta*(cum[t]/cum[tau]) * S[t][tau] * K[tau][i]
    for (int tt = wv; tt < 16; tt += 8) {
        int mt = tt >> 2, nt = tt & 3;
        int t = mt * 16 + r;
        float ct = cs[t];
        v8f acc = {};
        for (int k0 = 0; k0 < CT; k0 += 4) {
            v2f a, bb;
#pragma unroll
            for (int e = 0; e < 2; ++e) {
                int tau = k0 + e + 2 * hh;
                float w = (tau <= t) ? (ETA * ct / cs[tau]) : 0.0f;
                a[e]  = w * Ss[t][tau];
                bb[e] = Ks[tau][nt * 16 + r];
            }
            acc = wmma_f32x4(a, bb, acc);
        }
#pragma unroll
        for (int i = 0; i < 8; ++i) {
            int tg = mt * 16 + i + 8 * hh;
            size_t g = ((size_t)(b * Nseq + c * CT + tg)) * HD + h * DHd + nt * 16 + r;
            yws[g] = acc[i];
        }
    }

    // G[i][j] = sum_tau eta*(P/cum[tau]) * K[tau][i] * V[tau][j]
    for (int tt = wv; tt < 16; tt += 8) {
        int mt = tt >> 2, nt = tt & 3;
        v8f acc = {};
        for (int k0 = 0; k0 < CT; k0 += 4) {
            v2f a, bb;
#pragma unroll
            for (int e = 0; e < 2; ++e) {
                int tau = k0 + e + 2 * hh;
                a[e]  = (ETA * P / cs[tau]) * Ks[tau][mt * 16 + r];
                bb[e] = Vs[tau][nt * 16 + r];
            }
            acc = wmma_f32x4(a, bb, acc);
        }
        size_t base = (((size_t)(b * Hh + h)) * NC + c) * (DHd * DHd);
#pragma unroll
        for (int i = 0; i < 8; ++i)
            Gws[base + (size_t)(mt * 16 + i + 8 * hh) * DHd + nt * 16 + r] = acc[i];
    }
}

// ---------------- serial chunk scan: st = P*st + G ; Y_inter = (cum.Q) st^T ----
__global__ void __launch_bounds__(256) scan_kernel(
    const float* __restrict__ q, const float* __restrict__ state_in,
    const float* __restrict__ cum, const float* __restrict__ Gws,
    float* __restrict__ yws, float* __restrict__ state_out) {
    __shared__ float st[DHd][DHd];
    __shared__ float Qs[CT][DHd];
    __shared__ float cs[CT];
    int h = blockIdx.x, b = blockIdx.y;
    int lane = threadIdx.x & 31, wv = threadIdx.x >> 5;
    int r = lane & 15, hh = lane >> 4;

    const float* st_in = state_in + ((size_t)(b * Hh + h)) * (DHd * DHd);
    for (int i4 = threadIdx.x; i4 < (DHd * DHd) / 4; i4 += 256)
        ((float4*)&st[0][0])[i4] = ((const float4*)st_in)[i4];

    for (int c = 0; c < NC; ++c) {
        __syncthreads();
        const size_t rowbase = ((size_t)(b * Nseq + c * CT)) * HD + h * DHd;
        for (int i4 = threadIdx.x; i4 < (CT * DHd) / 4; i4 += 256) {
            int t = i4 >> 4, i = (i4 & 15) * 4;
            async_ld_b128(&Qs[t][i], q + rowbase + (size_t)t * HD + i);
        }
        if (threadIdx.x < CT) cs[threadIdx.x] = cum[b * Nseq + c * CT + threadIdx.x];
        async_wait0();
        __syncthreads();
        float P = cs[CT - 1];

        // Y_inter[t][i] = cum[t] * sum_j Q[t][j] * st[i][j]
        for (int tt = wv; tt < 16; tt += 8) {
            int mt = tt >> 2, nt = tt & 3;
            int t = mt * 16 + r;
            float ct = cs[t];
            v8f acc = {};
            for (int k0 = 0; k0 < DHd; k0 += 4) {
                v2f a, bb;
#pragma unroll
                for (int e = 0; e < 2; ++e) {
                    int j = k0 + e + 2 * hh;
                    a[e]  = ct * Qs[t][j];
                    bb[e] = st[nt * 16 + r][j];
                }
                acc = wmma_f32x4(a, bb, acc);
            }
#pragma unroll
            for (int i = 0; i < 8; ++i) {
                int tg = mt * 16 + i + 8 * hh;
                size_t g = ((size_t)(b * Nseq + c * CT + tg)) * HD + h * DHd + nt * 16 + r;
                yws[g] += acc[i];
            }
        }
        __syncthreads();
        const float* Gc = Gws + (((size_t)(b * Hh + h)) * NC + c) * (DHd * DHd);
        for (int i4 = threadIdx.x; i4 < (DHd * DHd) / 4; i4 += 256) {
            float4 g4 = ((const float4*)Gc)[i4];
            float4* sp = ((float4*)&st[0][0]) + i4;
            float4 s4 = *sp;
            s4.x = P * s4.x + g4.x;
            s4.y = P * s4.y + g4.y;
            s4.z = P * s4.z + g4.z;
            s4.w = P * s4.w + g4.w;
            *sp = s4;
        }
    }
    __syncthreads();
    float* st_o = state_out + ((size_t)(b * Hh + h)) * (DHd * DHd);
    for (int i4 = threadIdx.x; i4 < (DHd * DHd) / 4; i4 += 256)
        ((float4*)st_o)[i4] = ((float4*)&st[0][0])[i4];
}

// ---------------- output projection: out = y @ Wo + bo (bf16 WMMA) ------------
__global__ void __launch_bounds__(128) out_kernel(
    const __bf16* __restrict__ ybf, const __bf16* __restrict__ PWo,
    const float* __restrict__ bo, float* __restrict__ out) {
    int lane = threadIdx.x & 31, wv = threadIdx.x >> 5;
    int r = lane & 15, hh = lane >> 4;
    const int KB = HD >> 5;                          // 16
    int m  = blockIdx.x * 16 + r;
    int nt = blockIdx.y * 4 + wv;
    int col = nt * 16 + r;
    v8f acc = {};
    gemm_tile_bf16(ybf + (size_t)m * HD,
                   PWo + ((size_t)(nt * KB) * 32 + lane) * 16, KB, hh, acc);
#pragma unroll
    for (int i = 0; i < 8; ++i) {
        int mm = blockIdx.x * 16 + i + 8 * hh;
        out[(size_t)mm * Dm + col] = acc[i] + bo[col];
    }
}

extern "C" void kernel_launch(void* const* d_in, const int* in_sizes, int n_in,
                              void* d_out, int out_size, void* d_ws, size_t ws_size,
                              hipStream_t stream) {
    (void)in_sizes; (void)n_in; (void)out_size; (void)ws_size;
    const float* x     = (const float*)d_in[0];
    const float* gamma = (const float*)d_in[1];
    const float* state = (const float*)d_in[2];
    const float* Wq    = (const float*)d_in[3];
    const float* Wk    = (const float*)d_in[4];
    const float* Wv    = (const float*)d_in[5];
    const float* Wo    = (const float*)d_in[6];
    const float* bo    = (const float*)d_in[7];

    float* out    = (float*)d_out;
    float* st_out = out + (size_t)Bsz * Nseq * Dm;

    const size_t QKV = (size_t)Bsz * Nseq * HD;            // 4,194,304 floats
    float* qw   = (float*)d_ws;
    float* kw   = qw + QKV;
    float* vw   = kw + QKV;
    float* yw   = vw + QKV;
    float* Gw   = yw + QKV;
    float* cumw = Gw + QKV;
    __bf16* xb  = (__bf16*)(cumw + (size_t)Bsz * Nseq);
    __bf16* ybf = xb + (size_t)Bsz * Nseq * Dm;            // 8,388,608 bf16
    __bf16* PWq = ybf + QKV;                               // 4,194,304 bf16
    __bf16* PWk = PWq + (size_t)Dm * HD;
    __bf16* PWv = PWk + (size_t)Dm * HD;
    __bf16* PWo = PWv + (size_t)Dm * HD;

    decay_cum_kernel<<<dim3(1), dim3(128), 0, stream>>>(gamma, cumw);
    pack_bf16_kernel<<<dim3((Bsz * Nseq * Dm / 4) / 256), dim3(256), 0, stream>>>(
        x, xb, Bsz * Nseq * Dm / 4);
    pack_w_kernel<<<dim3(128), dim3(256), 0, stream>>>(Wq, PWq, Dm, HD);
    pack_w_kernel<<<dim3(128), dim3(256), 0, stream>>>(Wk, PWk, Dm, HD);
    pack_w_kernel<<<dim3(128), dim3(256), 0, stream>>>(Wv, PWv, Dm, HD);
    pack_w_kernel<<<dim3(128), dim3(256), 0, stream>>>(Wo, PWo, HD, Dm);

    proj_kernel<<<dim3((Bsz * Nseq) / 16, (HD / 16) / 4, 3), dim3(128), 0, stream>>>(
        xb, PWq, PWk, PWv, qw, kw, vw);
    chunk_intra_kernel<<<dim3(NC, Hh, Bsz), dim3(256), 0, stream>>>(qw, kw, vw, cumw, yw, Gw);
    scan_kernel<<<dim3(Hh, Bsz), dim3(256), 0, stream>>>(qw, state, cumw, Gw, yw, st_out);
    pack_bf16_kernel<<<dim3((Bsz * Nseq * HD / 4) / 256), dim3(256), 0, stream>>>(
        yw, ybf, Bsz * Nseq * HD / 4);
    out_kernel<<<dim3((Bsz * Nseq) / 16, (Dm / 16) / 4), dim3(128), 0, stream>>>(
        ybf, PWo, bo, out);
}
